// NeSSST_74698071212278
// MI455X (gfx1250) — matrix operand order repaired and checked by reading, not compiled, not run on hardware
//
#include <hip/hip_runtime.h>
#include <hip/hip_bf16.h>
#include <math.h>

typedef __attribute__((ext_vector_type(16))) _Float16 v16h;
typedef __attribute__((ext_vector_type(2)))  _Float16 v2h;
typedef __attribute__((ext_vector_type(8)))  float    v8f;

#define BATCH 4
#define IMH   512
#define IMW   512
#define KTOP  2048
#define MAXC  65536
#define EPSBN 1e-5f

// ---------------------------------------------------------------------------
// Weight repack: OIHW f32 -> f16 pairs in CDNA5 B-fragment order.
// K ordering: k = kyx*Cpad + cin  (Cpad = align32(Cin)), so every 32-wide K
// chunk has one uniform stencil tap.  Rows with cin>=Cin are zero -> the conv
// kernel never masks A for channel padding.
// out[(p*Npad + n)*2 + t] holds k = 2p+t, column n.
// ---------------------------------------------------------------------------
__global__ void pack_w_kernel(const float* __restrict__ w, _Float16* __restrict__ out,
                              int Cin, int Cpad, int Cout) {
  const int Npad = (Cout + 15) & ~15;
  const int Kpad = 9 * Cpad;
  const int total = (Kpad / 2) * Npad * 2;
  int i = blockIdx.x * blockDim.x + threadIdx.x;
  if (i >= total) return;
  int t    = i & 1;
  int rest = i >> 1;
  int n    = rest % Npad;
  int p    = rest / Npad;
  int k    = p * 2 + t;
  int kyx  = k / Cpad;
  int cin  = k - kyx * Cpad;
  float v = 0.0f;
  if (cin < Cin && n < Cout)
    v = w[((size_t)n * Cin + cin) * 9 + kyx];
  out[((size_t)p * Npad + n) * 2 + t] = (_Float16)v;
}

// ---------------------------------------------------------------------------
// Implicit-GEMM 3x3 SAME conv via V_WMMA_F32_16X16X32_F16.
// One wave -> 32 pixels (two adjacent 16-px x-strips) x 32 output channels
// (two 16x16 tiles): 4 WMMAs per chunk sharing 16 A-dwords + 16 B-dwords.
// All channel counts are even -> each K pair is one aligned b32 load, masked
// with a single cndmask.  Per chunk the stencil tap is scalar: row-OOB is a
// uniform chunk skip.  Dual-source input fuses the skip concat.
// outMode 1: single-channel f32 output clipped to [0,32] (final "ness" map).
// ---------------------------------------------------------------------------
__global__ __launch_bounds__(32)
void conv3x3_wmma_kernel(const _Float16* __restrict__ src0, int C0,
                         const _Float16* __restrict__ src1, int C1,
                         const _Float16* __restrict__ wpack,
                         const float* __restrict__ bias,
                         int Cin, int Cpad, int Cout, int Hh, int Ww,
                         _Float16* __restrict__ dst16,
                         float* __restrict__ dst32, int outMode) {
  const int spairs = Ww >> 5;                    // 32-pixel strip pairs per row
  const int pairI  = blockIdx.x / spairs;
  const int x0     = (blockIdx.x % spairs) << 5;
  const int y      = blockIdx.y;
  const int b      = blockIdx.z;
  const int lane   = threadIdx.x;
  const int m      = lane & 15;
  const int half   = lane >> 4;
  const int Npad   = (Cout + 15) & ~15;
  const int nG0    = pairI * 32 + m;
  const bool hasT1 = (pairI * 32 + 16) < Npad;   // wave-uniform
  const int xA     = x0 + m;                     // strip A pixel
  const int xB     = xA + 16;                    // strip B pixel
  const int CinE   = Cin - 2;                    // even clamp for pair starts

  const _Float16* s0 = src0 + (size_t)b * Hh * Ww * C0;
  const _Float16* s1 = src1 + (size_t)b * Hh * Ww * C1;

  v8f accA0 = {}, accA1 = {}, accB0 = {}, accB1 = {};

#pragma unroll
  for (int g = 0; g < 9; ++g) {
    const int ky = g / 3, kx = g - ky * 3;       // compile-time per unrolled g
    const int yy = y + ky - 1;
    if ((unsigned)yy >= (unsigned)Hh) continue;  // uniform: zero row contributes 0
    const size_t rowOff = (size_t)yy * Ww;

    const int xxA = xA + kx - 1;
    const bool okA = (unsigned)xxA < (unsigned)Ww;
    const int xcA = min(max(xxA, 0), Ww - 1);
    const unsigned* rA0 = (const unsigned*)(s0 + (rowOff + xcA) * C0);
    const unsigned* rA1 = (const unsigned*)(s1 + (rowOff + xcA) * C1);

    const int xxB = xB + kx - 1;
    const bool okB = (unsigned)xxB < (unsigned)Ww;
    const int xcB = min(max(xxB, 0), Ww - 1);
    const unsigned* rB0 = (const unsigned*)(s0 + (rowOff + xcB) * C0);
    const unsigned* rB1 = (const unsigned*)(s1 + (rowOff + xcB) * C1);

    for (int cb = 0; cb < Cpad; cb += 32) {
      if (cb + 32 < Cpad)
        __builtin_prefetch(wpack + ((size_t)((g * Cpad + cb + 32) >> 1)) * Npad * 2, 0, 1);
      v16h Aa, Ab, B0, B1;
#pragma unroll
      for (int j = 0; j < 8; ++j) {
        const int koff = 2 * j + 16 * half;
        int cc = cb + koff;                      // even pair start
        cc = (cc > CinE) ? CinE : cc;            // pad region: B rows are zero
        const bool inS0 = cc < C0;               // pairs never straddle the split
        const int h0 = cc >> 1, h1 = (cc - C0) >> 1;
        unsigned dwA = inS0 ? rA0[h0] : rA1[h1];
        unsigned dwB = inS0 ? rB0[h0] : rB1[h1];
        dwA = okA ? dwA : 0u;
        dwB = okB ? dwB : 0u;
        const v2h pA = __builtin_bit_cast(v2h, dwA);
        const v2h pB = __builtin_bit_cast(v2h, dwB);
        Aa[2 * j] = pA.x; Aa[2 * j + 1] = pA.y;
        Ab[2 * j] = pB.x; Ab[2 * j + 1] = pB.y;

        const int p = ((g * Cpad + cb) >> 1) + j + 8 * half;
        const _Float16* wb = wpack + ((size_t)p * Npad + nG0) * 2;
        B0[2 * j] = wb[0]; B0[2 * j + 1] = wb[1];
        if (hasT1) { B1[2 * j] = wb[32]; B1[2 * j + 1] = wb[33]; }
      }
      accA0 = __builtin_amdgcn_wmma_f32_16x16x32_f16(false, Aa, false, B0,
                                                     (short)0, accA0, false, false);
      accB0 = __builtin_amdgcn_wmma_f32_16x16x32_f16(false, Ab, false, B0,
                                                     (short)0, accB0, false, false);
      if (hasT1) {
        accA1 = __builtin_amdgcn_wmma_f32_16x16x32_f16(false, Aa, false, B1,
                                                       (short)0, accA1, false, false);
        accB1 = __builtin_amdgcn_wmma_f32_16x16x32_f16(false, Ab, false, B1,
                                                       (short)0, accB1, false, false);
      }
    }
  }

  const float bd0 = (nG0 < Cout) ? bias[nG0] : 0.0f;
  const float bd1 = (hasT1 && (nG0 + 16) < Cout) ? bias[nG0 + 16] : 0.0f;
#pragma unroll
  for (int r = 0; r < 8; ++r) {
    const int mo  = r + 8 * half;                // C/D layout: lanes16-31 hold M=r+8
    const int xpA = x0 + mo;
    const int xpB = x0 + 16 + mo;
    const size_t rowBase = ((size_t)b * Hh + y) * Ww;
    if (outMode == 0) {
      if (nG0 < Cout) {
        dst16[(rowBase + xpA) * Cout + nG0] = (_Float16)(accA0[r] + bd0);
        dst16[(rowBase + xpB) * Cout + nG0] = (_Float16)(accB0[r] + bd0);
      }
      if (hasT1 && (nG0 + 16) < Cout) {
        dst16[(rowBase + xpA) * Cout + nG0 + 16] = (_Float16)(accA1[r] + bd1);
        dst16[(rowBase + xpB) * Cout + nG0 + 16] = (_Float16)(accB1[r] + bd1);
      }
    } else {
      if (nG0 == 0) {
        dst32[rowBase + xpA] = fminf(fmaxf(accA0[r] + bd0, 0.0f), 32.0f);
        dst32[rowBase + xpB] = fminf(fmaxf(accB0[r] + bd0, 0.0f), 32.0f);
      }
    }
  }
}

// ---------------------------------------------------------------------------
// image NCHW f32 (3ch) -> NHWC f16 padded to 4 channels (ch 3 = 0) so the
// conv kernel's aligned-dword A path applies to every layer.
// ---------------------------------------------------------------------------
__global__ void cvt_image_kernel(const float* __restrict__ img, _Float16* __restrict__ out) {
  int i = blockIdx.x * blockDim.x + threadIdx.x;
  const int total = BATCH * IMH * IMW * 4;
  if (i >= total) return;
  int c = i % 4;
  int x = (i / 4) % IMW;
  int y = (i / (4 * IMW)) % IMH;
  int b = i / (4 * IMW * IMH);
  out[i] = (c < 3) ? (_Float16)img[(((size_t)b * 3 + c) * IMH + y) * IMW + x]
                   : (_Float16)0.0f;
}

// ---------------------------------------------------------------------------
// fused bn -> relu -> 2x2 maxpool, NHWC f16 -> NHWC f16 (half res)
// ---------------------------------------------------------------------------
__global__ void bnrelu_pool_kernel(const _Float16* __restrict__ in, _Float16* __restrict__ out,
                                   const float* __restrict__ g, const float* __restrict__ be,
                                   const float* __restrict__ mu, const float* __restrict__ va,
                                   int C, int Ho, int Wo) {
  int i = blockIdx.x * blockDim.x + threadIdx.x;
  const int total = BATCH * Ho * Wo * C;
  if (i >= total) return;
  int c = i % C;
  int x = (i / C) % Wo;
  int y = (i / (C * Wo)) % Ho;
  int b = i / (C * Wo * Ho);
  const int Hi = Ho * 2, Wi = Wo * 2;
  const float inv = g[c] * rsqrtf(va[c] + EPSBN);
  const float sh  = be[c] - mu[c] * inv;
  float mv = -1e30f;
#pragma unroll
  for (int dy = 0; dy < 2; ++dy)
#pragma unroll
    for (int dx = 0; dx < 2; ++dx) {
      float v = (float)in[(((size_t)b * Hi + (2 * y + dy)) * Wi + (2 * x + dx)) * C + c];
      v = fmaxf(v * inv + sh, 0.0f);
      mv = fmaxf(mv, v);
    }
  out[i] = (_Float16)mv;
}

// ---------------------------------------------------------------------------
// fused bn -> relu -> 2x2 stride-2 transposed conv; bn scale/shift staged in LDS
// ---------------------------------------------------------------------------
__global__ __launch_bounds__(256)
void bnrelu_upconv_kernel(const _Float16* __restrict__ xin, _Float16* __restrict__ xu,
                          const float* __restrict__ g, const float* __restrict__ be,
                          const float* __restrict__ mu, const float* __restrict__ va,
                          const float* __restrict__ w, const float* __restrict__ ub,
                          int Ci, int Ho, int Wo) {
  __shared__ float sInv[128], sSh[128];
  for (int c = threadIdx.x; c < Ci; c += 256) {
    const float inv = g[c] * rsqrtf(va[c] + EPSBN);
    sInv[c] = inv;
    sSh[c]  = be[c] - mu[c] * inv;
  }
  __syncthreads();
  int i = blockIdx.x * blockDim.x + threadIdx.x;
  const int total = BATCH * Ho * Wo * Ci;
  if (i >= total) return;
  int co = i % Ci;
  int X  = (i / Ci) % Wo;
  int Y  = (i / (Ci * Wo)) % Ho;
  int b  = i / (Ci * Wo * Ho);
  const int hi = Ho >> 1, wi = Wo >> 1;
  const int yi = Y >> 1, xi = X >> 1;
  const int tap = ((Y & 1) << 1) | (X & 1);
  const _Float16* px = xin + (((size_t)b * hi + yi) * wi + xi) * Ci;
  float s = ub[co];
  for (int ci = 0; ci < Ci; ++ci) {
    const float v = fmaxf((float)px[ci] * sInv[ci] + sSh[ci], 0.0f);
    s = fmaf(v, w[((size_t)co * Ci + ci) * 4 + tap], s);
  }
  xu[i] = (_Float16)s;
}

// ---------------------------------------------------------------------------
// Shi-Tomasi part 1: gradients of gray, store dx*dx, dy*dy, dx*dy
// ---------------------------------------------------------------------------
__device__ __forceinline__ float grayRead(const float* g, int b, int y, int x) {
  if ((unsigned)y >= (unsigned)IMH || (unsigned)x >= (unsigned)IMW) return 0.0f;
  return g[((size_t)b * IMH + y) * IMW + x];
}

__global__ void shi_grad_kernel(const float* __restrict__ gray,
                                float* __restrict__ pxx, float* __restrict__ pyy,
                                float* __restrict__ pxy) {
  int i = blockIdx.x * blockDim.x + threadIdx.x;
  const int total = BATCH * IMH * IMW;
  if (i >= total) return;
  int x = i % IMW;
  int y = (i / IMW) % IMH;
  int b = i / (IMW * IMH);
  float dx = -0.5f * grayRead(gray, b, y - 1, x - 1) + 0.5f * grayRead(gray, b, y - 1, x + 1)
             - 1.0f * grayRead(gray, b, y,     x - 1) + 1.0f * grayRead(gray, b, y,     x + 1)
             - 0.5f * grayRead(gray, b, y + 1, x - 1) + 0.5f * grayRead(gray, b, y + 1, x + 1);
  float dy = -0.5f * grayRead(gray, b, y - 1, x - 1) + 0.5f * grayRead(gray, b, y + 1, x - 1)
             - 1.0f * grayRead(gray, b, y - 1, x)     + 1.0f * grayRead(gray, b, y + 1, x)
             - 0.5f * grayRead(gray, b, y - 1, x + 1) + 0.5f * grayRead(gray, b, y + 1, x + 1);
  pxx[i] = dx * dx;
  pyy[i] = dy * dy;
  pxy[i] = dx * dy;
}

// ---------------------------------------------------------------------------
// Shi-Tomasi part 2: gaussian-window second moments -> min-eigenvalue score
// ---------------------------------------------------------------------------
struct GaussW { float w[9]; };

__device__ __forceinline__ float mapRead(const float* p, int b, int y, int x) {
  if ((unsigned)y >= (unsigned)IMH || (unsigned)x >= (unsigned)IMW) return 0.0f;
  return p[((size_t)b * IMH + y) * IMW + x];
}

__global__ void shi_score_kernel(const float* __restrict__ pxx, const float* __restrict__ pyy,
                                 const float* __restrict__ pxy, float* __restrict__ score,
                                 GaussW gw) {
  int i = blockIdx.x * blockDim.x + threadIdx.x;
  const int total = BATCH * IMH * IMW;
  if (i >= total) return;
  int x = i % IMW;
  int y = (i / IMW) % IMH;
  int b = i / (IMW * IMH);
  float a = 0.0f, bm = 0.0f, c = 0.0f;
#pragma unroll
  for (int ky = 0; ky < 3; ++ky)
#pragma unroll
    for (int kx = 0; kx < 3; ++kx) {
      const float wg = gw.w[ky * 3 + kx];
      a  = fmaf(wg, mapRead(pyy, b, y + ky - 1, x + kx - 1), a);
      bm = fmaf(wg, mapRead(pxy, b, y + ky - 1, x + kx - 1), bm);
      c  = fmaf(wg, mapRead(pxx, b, y + ky - 1, x + kx - 1), c);
    }
  const float tr  = a + c;
  const float det = a * c - bm * bm;
  const float d   = tr * tr - 4.0f * det;
  const float mask = (d >= 0.0f) ? 1.0f : 0.0f;
  const float sq = sqrtf(d * mask);
  score[i] = (tr - sq) * 0.5f * mask;
}

// ---------------------------------------------------------------------------
// NMS mask * exp(-ness) * border mask  ->  selection score "s"
// ---------------------------------------------------------------------------
__global__ void smap_kernel(const float* __restrict__ score, const float* __restrict__ ness,
                            const int* __restrict__ nmsPtr, float* __restrict__ smap) {
  int i = blockIdx.x * blockDim.x + threadIdx.x;
  const int total = BATCH * IMH * IMW;
  if (i >= total) return;
  int x = i % IMW;
  int y = (i / IMW) % IMH;
  int b = i / (IMW * IMH);
  const int pad = (*nmsPtr) >> 1;
  const float sc = score[i];
  float mx = -1e30f;
  for (int dy = -pad; dy <= pad; ++dy) {
    const int yy = y + dy;
    if ((unsigned)yy >= (unsigned)IMH) continue;
    for (int dx = -pad; dx <= pad; ++dx) {
      const int xx = x + dx;
      if ((unsigned)xx >= (unsigned)IMW) continue;
      mx = fmaxf(mx, score[((size_t)b * IMH + yy) * IMW + xx]);
    }
  }
  const bool m = (sc == mx) && (sc > 0.0f);
  const bool inb = (y >= 2) && (y < IMH - 2) && (x >= 2) && (x < IMW - 2);
  smap[i] = (m && inb) ? expf(-ness[i]) : 0.0f;
}

// ---------------------------------------------------------------------------
// Top-k: compact candidates, exact-rank selection (value desc, index asc)
// ---------------------------------------------------------------------------
__global__ void zero_cnt_kernel(int* cnt) {
  int i = blockIdx.x * blockDim.x + threadIdx.x;
  if (i < BATCH) cnt[i] = 0;
}

__global__ void compact_kernel(const float* __restrict__ smap,
                               unsigned long long* __restrict__ cand, int* __restrict__ cnt) {
  int i = blockIdx.x * blockDim.x + threadIdx.x;
  const int total = BATCH * IMH * IMW;
  if (i >= total) return;
  const int pix = i % (IMH * IMW);
  const int b = i / (IMH * IMW);
  const float v = smap[i];
  if (v > 0.0f) {
    const int pos = atomicAdd(&cnt[b], 1);
    if (pos < MAXC) {
      const unsigned bits = __float_as_uint(v);   // v > 0 -> bit order == value order
      const unsigned long long key =
          ((unsigned long long)bits << 32) | (unsigned)(~(unsigned)pix);
      cand[(size_t)b * MAXC + pos] = key;
    }
  }
}

__global__ __launch_bounds__(256)
void rank_select_kernel(const unsigned long long* __restrict__ cand, const int* __restrict__ cnt,
                        float* __restrict__ kp) {
  __shared__ unsigned long long sk[256];
  const int b = blockIdx.y;
  int nc = cnt[b];
  if (nc > MAXC) nc = MAXC;
  const int tid = threadIdx.x;
  const int i = blockIdx.x * 256 + tid;
  const unsigned long long mykey = (i < nc) ? cand[(size_t)b * MAXC + i] : 0ull;
  int rank = 0;
  for (int base = 0; base < nc; base += 256) {
    const int jj = base + tid;
    sk[tid] = (jj < nc) ? cand[(size_t)b * MAXC + jj] : 0ull;
    __syncthreads();
    const int lim = min(256, nc - base);
    if (i < nc)
      for (int j = 0; j < lim; ++j) rank += (sk[j] > mykey) ? 1 : 0;
    __syncthreads();
  }
  if (i < nc && rank < KTOP) {
    const unsigned flat = ~(unsigned)(mykey & 0xffffffffull);
    const int yy = flat / IMW;
    const int xx = flat - yy * IMW;
    kp[((size_t)b * KTOP + rank) * 2 + 0] = (float)yy + 0.5f;
    kp[((size_t)b * KTOP + rank) * 2 + 1] = (float)xx + 0.5f;
  }
}

__global__ void fill_rest_kernel(const int* __restrict__ cnt, float* __restrict__ kp) {
  int i = blockIdx.x * blockDim.x + threadIdx.x;
  if (i >= BATCH * KTOP) return;
  const int r = i % KTOP;
  const int b = i / KTOP;
  int base = cnt[b];
  if (base > KTOP) base = KTOP;
  if (r >= base) {
    const unsigned flat = (unsigned)(r - base);   // ties at 0.0 -> ascending index
    const int yy = flat / IMW;
    const int xx = flat - yy * IMW;
    kp[((size_t)b * KTOP + r) * 2 + 0] = (float)yy + 0.5f;
    kp[((size_t)b * KTOP + r) * 2 + 1] = (float)xx + 0.5f;
  }
}

// ---------------------------------------------------------------------------
// Subpixel localization: bilinear-sampled score derivatives + Newton step
// ---------------------------------------------------------------------------
__device__ __forceinline__ float Srd(const float* s, int b, int y, int x) {
  if ((unsigned)y >= (unsigned)IMH || (unsigned)x >= (unsigned)IMW) return 0.0f;
  return s[((size_t)b * IMH + y) * IMW + x];
}
__device__ __forceinline__ float dDX(const float* s, int b, int y, int x) {
  return 0.5f * (Srd(s, b, y, x + 1) - Srd(s, b, y, x - 1));
}
__device__ __forceinline__ float dDY(const float* s, int b, int y, int x) {
  return 0.5f * (Srd(s, b, y + 1, x) - Srd(s, b, y - 1, x));
}
__device__ __forceinline__ float dDXDX(const float* s, int b, int y, int x) {
  return Srd(s, b, y, x - 1) - 2.0f * Srd(s, b, y, x) + Srd(s, b, y, x + 1);
}
__device__ __forceinline__ float dDYDY(const float* s, int b, int y, int x) {
  return Srd(s, b, y - 1, x) - 2.0f * Srd(s, b, y, x) + Srd(s, b, y + 1, x);
}
__device__ __forceinline__ float dDXDY(const float* s, int b, int y, int x) {
  return 0.25f * (Srd(s, b, y - 1, x - 1) - Srd(s, b, y - 1, x + 1)
                - Srd(s, b, y + 1, x - 1) + Srd(s, b, y + 1, x + 1));
}

__global__ void localize_kernel(const float* __restrict__ score, const float* __restrict__ kp,
                                float* __restrict__ out) {
  int i = blockIdx.x * blockDim.x + threadIdx.x;
  if (i >= BATCH * KTOP) return;
  const int b = i / KTOP;
  const float kpy = kp[(size_t)i * 2 + 0];
  const float kpx = kp[(size_t)i * 2 + 1];
  const float py = kpy - 0.5f, px = kpx - 0.5f;
  const float y0f = floorf(py), x0f = floorf(px);
  const float wy = py - y0f, wx = px - x0f;
  const int y0 = (int)y0f, x0 = (int)x0f;
  const float ww[2][2] = { { (1.f - wy) * (1.f - wx), (1.f - wy) * wx },
                           { wy * (1.f - wx), wy * wx } };
  float kdx = 0.f, kdy = 0.f, kxx = 0.f, kyy = 0.f, kxy = 0.f;
#pragma unroll
  for (int iy = 0; iy < 2; ++iy)
#pragma unroll
    for (int ix = 0; ix < 2; ++ix) {
      const int yy = y0 + iy, xx = x0 + ix;
      const float vmask =
          (yy >= 0 && yy < IMH && xx >= 0 && xx < IMW) ? 1.0f : 0.0f;
      const int yc = min(max(yy, 0), IMH - 1);
      const int xc = min(max(xx, 0), IMW - 1);
      const float wgt = ww[iy][ix] * vmask;
      if (wgt != 0.0f) {
        kdx = fmaf(wgt, dDX(score, b, yc, xc), kdx);
        kdy = fmaf(wgt, dDY(score, b, yc, xc), kdy);
        kxx = fmaf(wgt, dDXDX(score, b, yc, xc), kxx);
        kyy = fmaf(wgt, dDYDY(score, b, yc, xc), kyy);
        kxy = fmaf(wgt, dDXDY(score, b, yc, xc), kxy);
      }
    }
  const float det = fmaxf(kxx * kyy - kxy * kxy, 1e-8f);
  const float ok = (det > 1e-3f) ? 1.0f : 0.0f;
  const float i00 = kxx / det, i01 = -kxy / det, i11 = kyy / det;
  const float locy = -(i00 * kdy + i01 * kdx);
  const float locx = -(i01 * kdy + i11 * kdx);
  out[(size_t)i * 2 + 0] = kpy + locy * ok;
  out[(size_t)i * 2 + 1] = kpx + locx * ok;
}

// ---------------------------------------------------------------------------
// host orchestration
// ---------------------------------------------------------------------------
static inline int ceil_div(long long a, int b) { return (int)((a + b - 1) / b); }

extern "C" void kernel_launch(void* const* d_in, const int* in_sizes, int n_in,
                              void* d_out, int out_size, void* d_ws, size_t ws_size,
                              hipStream_t stream) {
  (void)in_sizes; (void)n_in; (void)out_size; (void)ws_size;
  const int B = BATCH, H = IMH, W = IMW;

  // ---- input index map (insertion-order flatten of setup_inputs) ----
  // 0 image, 1 image_gray, 2.. params, 78 nms_size, 79 k
  static const int convW[18]   = {2,4,10,12,18,20,26,28,34,36,44,46,54,56,64,66,74,76};
  static const int convB[18]   = {3,5,11,13,19,21,27,29,35,37,45,47,55,57,65,67,75,77};
  static const int packCin[18] = {3,16,16,32,32,64,64,64,64,64,128,128,128,128,96,96,80,80};
  static const int convCin[18] = {4,16,16,32,32,64,64,64,64,64,128,128,128,128,96,96,80,80};
  static const int convCout[18]= {16,16,32,32,64,64,64,64,64,64,128,64,128,64,96,64,80,1};
  static const int bnIdx[8] = {6,14,22,30, 38,48,58,68};   // d1..d4, u0..u3
  static const int upW[4] = {42,52,62,72};
  static const int upB[4] = {43,53,63,73};

  // ---- workspace carve-out ----
  size_t off = 0;
  auto alloc = [&](size_t bytes) -> char* {
    off = (off + 255) & ~(size_t)255;
    char* p = (char*)d_ws + off;
    off += bytes;
    return p;
  };
  _Float16* t_img = (_Float16*)alloc((size_t)B * H * W * 4 * 2);
  _Float16* feat0 = (_Float16*)alloc((size_t)B * 512 * 512 * 16 * 2);
  _Float16* feat1 = (_Float16*)alloc((size_t)B * 256 * 256 * 32 * 2);
  _Float16* feat2 = (_Float16*)alloc((size_t)B * 128 * 128 * 64 * 2);
  _Float16* feat3 = (_Float16*)alloc((size_t)B * 64 * 64 * 64 * 2);
  _Float16* x4    = (_Float16*)alloc((size_t)B * 32 * 32 * 64 * 2);
  _Float16* poolbuf = (_Float16*)alloc((size_t)B * 256 * 256 * 16 * 2);
  _Float16* c0buf = (_Float16*)alloc((size_t)B * 512 * 512 * 80 * 2);
  _Float16* xubuf = (_Float16*)alloc((size_t)B * 512 * 512 * 64 * 2);
  _Float16* ubuf  = (_Float16*)alloc((size_t)B * 256 * 256 * 64 * 2);
  float* ness  = (float*)alloc((size_t)B * H * W * 4);
  float* pxx   = (float*)alloc((size_t)B * H * W * 4);
  float* pyy   = (float*)alloc((size_t)B * H * W * 4);
  float* pxy   = (float*)alloc((size_t)B * H * W * 4);
  float* score = (float*)alloc((size_t)B * H * W * 4);
  float* smap  = (float*)alloc((size_t)B * H * W * 4);
  _Float16* wpackBase = (_Float16*)alloc((size_t)8 << 20);
  unsigned long long* cand = (unsigned long long*)alloc((size_t)B * MAXC * 8);
  int* cnt = (int*)alloc(64);
  float* kpbuf = (float*)alloc((size_t)B * KTOP * 2 * 4);

  // ---- pack all conv weights to f16 B-fragment layout ----
  _Float16* wptr[18];
  {
    size_t wo = 0;
    for (int li = 0; li < 18; ++li) {
      const int Cin = packCin[li], Cout = convCout[li];
      const int Cpad = (convCin[li] + 31) & ~31;
      const int Npad = (Cout + 15) & ~15;
      const int Kpad = 9 * Cpad;
      wptr[li] = wpackBase + wo;
      wo += (size_t)(Kpad / 2) * Npad * 2;
      const int total = (Kpad / 2) * Npad * 2;
      pack_w_kernel<<<ceil_div(total, 256), 256, 0, stream>>>(
          (const float*)d_in[convW[li]], wptr[li], Cin, Cpad, Cout);
    }
  }

  // ---- image to NHWC f16 (4ch, zero-padded) ----
  cvt_image_kernel<<<ceil_div((long long)B * H * W * 4, 256), 256, 0, stream>>>(
      (const float*)d_in[0], t_img);

  auto conv = [&](int li, const _Float16* s0, int C0, const _Float16* s1, int C1,
                  int Hh, int Ww, _Float16* d16, float* d32, int mode) {
    const int Cout = convCout[li];
    const int Npad = (Cout + 15) & ~15;
    const int pairs = (Npad / 16 + 1) / 2;
    const int Cpad = (convCin[li] + 31) & ~31;
    dim3 g((Ww / 32) * pairs, Hh, B);
    conv3x3_wmma_kernel<<<g, 32, 0, stream>>>(
        s0, C0, (s1 ? s1 : s0), C1, wptr[li], (const float*)d_in[convB[li]],
        convCin[li], Cpad, Cout, Hh, Ww, d16, d32, mode);
  };
  auto bnpool = [&](int bi, const _Float16* in, _Float16* out, int C, int Ho, int Wo) {
    const int g0 = bnIdx[bi];
    bnrelu_pool_kernel<<<ceil_div((long long)B * Ho * Wo * C, 256), 256, 0, stream>>>(
        in, out, (const float*)d_in[g0], (const float*)d_in[g0 + 1],
        (const float*)d_in[g0 + 2], (const float*)d_in[g0 + 3], C, Ho, Wo);
  };
  auto upconv = [&](int ui, const _Float16* in, _Float16* out, int Ci, int Ho, int Wo) {
    const int g0 = bnIdx[4 + ui];
    bnrelu_upconv_kernel<<<ceil_div((long long)B * Ho * Wo * Ci, 256), 256, 0, stream>>>(
        in, out, (const float*)d_in[g0], (const float*)d_in[g0 + 1],
        (const float*)d_in[g0 + 2], (const float*)d_in[g0 + 3],
        (const float*)d_in[upW[ui]], (const float*)d_in[upB[ui]], Ci, Ho, Wo);
  };

  // ---- UNet encoder ----
  conv(0, t_img, 4, nullptr, 0, 512, 512, c0buf, nullptr, 0);
  conv(1, c0buf, 16, nullptr, 0, 512, 512, feat0, nullptr, 0);
  bnpool(0, feat0, poolbuf, 16, 256, 256);
  conv(2, poolbuf, 16, nullptr, 0, 256, 256, c0buf, nullptr, 0);
  conv(3, c0buf, 32, nullptr, 0, 256, 256, feat1, nullptr, 0);
  bnpool(1, feat1, poolbuf, 32, 128, 128);
  conv(4, poolbuf, 32, nullptr, 0, 128, 128, c0buf, nullptr, 0);
  conv(5, c0buf, 64, nullptr, 0, 128, 128, feat2, nullptr, 0);
  bnpool(2, feat2, poolbuf, 64, 64, 64);
  conv(6, poolbuf, 64, nullptr, 0, 64, 64, c0buf, nullptr, 0);
  conv(7, c0buf, 64, nullptr, 0, 64, 64, feat3, nullptr, 0);
  bnpool(3, feat3, poolbuf, 64, 32, 32);
  conv(8, poolbuf, 64, nullptr, 0, 32, 32, c0buf, nullptr, 0);
  conv(9, c0buf, 64, nullptr, 0, 32, 32, x4, nullptr, 0);

  // ---- UNet decoder (concat fused into conv A-gather) ----
  upconv(0, x4, xubuf, 64, 64, 64);
  conv(10, xubuf, 64, feat3, 64, 64, 64, c0buf, nullptr, 0);
  conv(11, c0buf, 128, nullptr, 0, 64, 64, ubuf, nullptr, 0);
  upconv(1, ubuf, xubuf, 64, 128, 128);
  conv(12, xubuf, 64, feat2, 64, 128, 128, c0buf, nullptr, 0);
  conv(13, c0buf, 128, nullptr, 0, 128, 128, ubuf, nullptr, 0);
  upconv(2, ubuf, xubuf, 64, 256, 256);
  conv(14, xubuf, 64, feat1, 32, 256, 256, c0buf, nullptr, 0);
  conv(15, c0buf, 96, nullptr, 0, 256, 256, ubuf, nullptr, 0);
  upconv(3, ubuf, xubuf, 64, 512, 512);
  conv(16, xubuf, 64, feat0, 16, 512, 512, c0buf, nullptr, 0);
  conv(17, c0buf, 80, nullptr, 0, 512, 512, nullptr, ness, 1);   // clipped ness f32

  // ---- Shi-Tomasi score ----
  shi_grad_kernel<<<ceil_div((long long)B * H * W, 256), 256, 0, stream>>>(
      (const float*)d_in[1], pxx, pyy, pxy);
  GaussW gw;
  {
    float c[3] = {-1.0f, 0.0f, 1.0f};
    float sum = 0.0f;
    for (int i = 0; i < 3; ++i)
      for (int j = 0; j < 3; ++j) {
        float d2 = c[i] * c[i] + c[j] * c[j];
        gw.w[i * 3 + j] = expf(-0.5f * d2 / 2.0f);
        sum += gw.w[i * 3 + j];
      }
    for (int i = 0; i < 9; ++i) gw.w[i] /= sum;
  }
  shi_score_kernel<<<ceil_div((long long)B * H * W, 256), 256, 0, stream>>>(
      pxx, pyy, pxy, score, gw);

  // ---- NMS + selection score ----
  smap_kernel<<<ceil_div((long long)B * H * W, 256), 256, 0, stream>>>(
      score, ness, (const int*)d_in[78], smap);

  // ---- exact top-k by rank ----
  zero_cnt_kernel<<<1, 32, 0, stream>>>(cnt);
  compact_kernel<<<ceil_div((long long)B * H * W, 256), 256, 0, stream>>>(smap, cand, cnt);
  {
    dim3 g(MAXC / 256, B);
    rank_select_kernel<<<g, 256, 0, stream>>>(cand, cnt, kpbuf);
  }
  fill_rest_kernel<<<ceil_div((long long)B * KTOP, 256), 256, 0, stream>>>(cnt, kpbuf);

  // ---- subpixel localization -> output (B, k, 2) ----
  localize_kernel<<<ceil_div((long long)B * KTOP, 256), 256, 0, stream>>>(
      score, kpbuf, (float*)d_out);
}
